// DynamLinear_15101105013222
// MI455X (gfx1250) — matrix operand order repaired and checked by reading, not compiled, not run on hardware
//
#include <hip/hip_runtime.h>
#include <math.h>

// CDNA5 (gfx1250) WMMA vector types
typedef __bf16 v16bf __attribute__((ext_vector_type(16)));
typedef __bf16 v8bf  __attribute__((ext_vector_type(8)));
typedef __bf16 v4bf  __attribute__((ext_vector_type(4)));
typedef float  v8f   __attribute__((ext_vector_type(8)));
typedef float  v4f   __attribute__((ext_vector_type(4)));

#define FEAT     1024
#define NCB      16              // codebooks (h)
#define OWID     64              // sum(CODE_DIMS)
#define CD       32              // each code dim
#define TOK_TILE 16              // tokens per workgroup (M of WMMA tile)
#define BLOCK    512             // 16 waves of 32
#define KB_STEPS (FEAT / 32)     // 32 K-blocks of 32

// Preprocessed weights: [h][kb][col][khalf] -> 16 contiguous bf16 (one B half-fragment)
#define WPREP_ELEMS (NCB * KB_STEPS * OWID * 2 * 16)      // 1,048,576 bf16 per plane
#define WPREP_BYTES ((size_t)2 * WPREP_ELEMS * 2)         // hi + lo planes = 4 MB
#define KB_BYTES    (OWID * 2 * 16 * 2)                   // 4096 B advance per kb
#define SUB_BYTES   (16 * 2 * 16 * 2)                     // 1024 B advance per subtile

__device__ __forceinline__ void bf16_split(float v, __bf16& hi, __bf16& lo) {
    hi = (__bf16)v;
    lo = (__bf16)(v - (float)hi);
}

// ---------------------------------------------------------------------------
// Prep: fp32 weights -> bf16 hi/lo planes in B-fragment order.
// Fragment index tid = ((h*32 + kb)*64 + col)*2 + khalf ; each thread emits 16 bf16.
// ---------------------------------------------------------------------------
__global__ __launch_bounds__(256)
void prep_weights(const float* __restrict__ w,
                  __bf16* __restrict__ whi, __bf16* __restrict__ wlo)
{
    const int tid   = blockIdx.x * 256 + threadIdx.x;   // 65536 threads
    const int khalf = tid & 1;
    const int col   = (tid >> 1) & 63;
    const int kb    = (tid >> 7) & 31;
    const int h     = tid >> 12;

    const float* __restrict__ src =
        w + (size_t)h * FEAT * OWID + (size_t)(kb * 32 + khalf * 16) * OWID + col;

    const size_t base = (size_t)tid * 16;
    #pragma unroll
    for (int e = 0; e < 16; ++e) {
        __bf16 hv, lv;
        bf16_split(src[e * OWID], hv, lv);
        whi[base + e] = hv;
        wlo[base + e] = lv;
    }
}

// ---------------------------------------------------------------------------
// Fused: WMMA GEMM (bf16 x3 split precision) + outer product + mean + rmsnorm
// ---------------------------------------------------------------------------
template<bool PREP>
__global__ __launch_bounds__(BLOCK)
void dynam_linear_fused(const float* __restrict__ x,
                        const float* __restrict__ w,
                        const __bf16* __restrict__ whi,
                        const __bf16* __restrict__ wlo,
                        float* __restrict__ out)
{
    // 64KB LDS: bf16 hi/lo x-stage during GEMM, then reused as f32 y-stage.
    __shared__ __align__(16) unsigned char lds_raw[TOK_TILE * FEAT * 4];
    __bf16* xs_hi = (__bf16*)lds_raw;                            // 32KB
    __bf16* xs_lo = (__bf16*)(lds_raw + TOK_TILE * FEAT * 2);    // 32KB
    float*  ys    = (float*)lds_raw;                             // 64KB (after GEMM)

    const int tid  = threadIdx.x;
    const int wave = tid >> 5;          // codebook h (GEMM) / token (epilogue)
    const int lane = tid & 31;
    const int t0   = blockIdx.x * TOK_TILE;

    // ---- Stage x tile: non-temporal coalesced float4 loads, split once ----
    {
        const v4f* __restrict__ xin =
            reinterpret_cast<const v4f*>(x + (size_t)t0 * FEAT);
        #pragma unroll
        for (int i = 0; i < (TOK_TILE * FEAT / 4) / BLOCK; ++i) {
            const int q = tid + i * BLOCK;
            const v4f v = __builtin_nontemporal_load(xin + q);   // read-once stream
            v4bf hi, lo;
            #pragma unroll
            for (int j = 0; j < 4; ++j) {
                __bf16 hv, lv;
                bf16_split(v[j], hv, lv);
                hi[j] = hv; lo[j] = lv;
            }
            *reinterpret_cast<v4bf*>(xs_hi + (size_t)q * 4) = hi;
            *reinterpret_cast<v4bf*>(xs_lo + (size_t)q * 4) = lo;
        }
    }
    __syncthreads();

    // ---- WMMA GEMM: y[16 tok x 64] for codebook h = wave ----
    const int  h     = wave;
    const bool hi16  = (lane >= 16);
    const int  mrow  = lane & 15;
    // A (16x32) k order used here (contiguous, same as B): lanes 0-15 hold K 0..15, lanes 16-31 hold K 16..31
    const int  ag0   = hi16 ? 16 : 0;
    const int  ag1   = hi16 ? 24 : 8;
    // B (32x16) k order used here (contiguous, same as A): lanes 0-15 hold K 0..15, lanes 16-31 hold K 16..31
    const int  khalf = hi16 ? 1 : 0;

    v8f acc[4];
    #pragma unroll
    for (int s = 0; s < 4; ++s)
        #pragma unroll
        for (int r = 0; r < 8; ++r) acc[s][r] = 0.0f;

    // 32-bit per-lane byte offset into each prep plane (kb=0, subtile 0);
    // subtile stride and fragment-half fold into the load immediate offset.
    unsigned boff = (((unsigned)h * KB_STEPS * OWID + (unsigned)mrow) * 2 +
                     (unsigned)khalf) * 32;
    const unsigned char* __restrict__ whiB = (const unsigned char*)whi;
    const unsigned char* __restrict__ wloB = (const unsigned char*)wlo;

    for (int kb = 0; kb < KB_STEPS; ++kb) {
        // A fragments: 4x ds_load_b128, zero conversion math
        const __bf16* arow_h = xs_hi + mrow * FEAT + kb * 32;
        const __bf16* arow_l = xs_lo + mrow * FEAT + kb * 32;
        const v8bf ah0 = *reinterpret_cast<const v8bf*>(arow_h + ag0);
        const v8bf ah1 = *reinterpret_cast<const v8bf*>(arow_h + ag1);
        const v8bf al0 = *reinterpret_cast<const v8bf*>(arow_l + ag0);
        const v8bf al1 = *reinterpret_cast<const v8bf*>(arow_l + ag1);
        const v16bf a_hi = __builtin_shufflevector(ah0, ah1, 0,1,2,3,4,5,6,7,8,9,10,11,12,13,14,15);
        const v16bf a_lo = __builtin_shufflevector(al0, al1, 0,1,2,3,4,5,6,7,8,9,10,11,12,13,14,15);

        #pragma unroll
        for (int s = 0; s < 4; ++s) {
            v16bf b_hi, b_lo;
            if constexpr (PREP) {
                // contiguous 32B fragments; s*SUB_BYTES folds into IOFFSET
                b_hi = *reinterpret_cast<const v16bf*>(whiB + boff + s * SUB_BYTES);
                b_lo = *reinterpret_cast<const v16bf*>(wloB + boff + s * SUB_BYTES);
            } else {
                const int col = s * 16 + mrow;
                const float* __restrict__ wcol =
                    w + (size_t)h * FEAT * OWID +
                    (size_t)(kb * 32 + khalf * 16) * OWID + col;
                #pragma unroll
                for (int e = 0; e < 16; ++e) {
                    __bf16 hv, lv;
                    bf16_split(wcol[e * OWID], hv, lv);
                    b_hi[e] = hv; b_lo[e] = lv;
                }
            }
            // bf16 x3 split-precision accumulation (drop lo*lo term)
            acc[s] = __builtin_amdgcn_wmma_f32_16x16x32_bf16(false, a_hi, false, b_hi, (short)0, acc[s], false, false);
            acc[s] = __builtin_amdgcn_wmma_f32_16x16x32_bf16(false, a_lo, false, b_hi, (short)0, acc[s], false, false);
            acc[s] = __builtin_amdgcn_wmma_f32_16x16x32_bf16(false, a_hi, false, b_lo, (short)0, acc[s], false, false);
        }
        boff += KB_BYTES;
    }
    __syncthreads();   // all waves done reading x-stage

    // ---- Scatter y tile into LDS: ys[tok][h*64 + n] ----
    // C/D layout: VGPR r -> token = r + (lane>=16 ? 8 : 0), N = lane & 15
    #pragma unroll
    for (int s = 0; s < 4; ++s) {
        const int n = s * 16 + mrow;
        #pragma unroll
        for (int r = 0; r < 8; ++r) {
            const int tokr = r + (hi16 ? 8 : 0);
            ys[tokr * FEAT + h * OWID + n] = acc[s][r];
        }
    }
    __syncthreads();

    // ---- Outer product + mean + rmsnorm: wave handles one token ----
    {
        const int tok = wave;
        const float* yrow = ys + tok * FEAT;
        float ov[CD];
        #pragma unroll
        for (int j = 0; j < CD; ++j) ov[j] = 0.0f;

        #pragma unroll
        for (int hh = 0; hh < NCB; ++hh) {
            const float av = yrow[hh * OWID + lane];     // a[hh, d1=lane]
            const v4f* bp4 = reinterpret_cast<const v4f*>(yrow + hh * OWID + CD);
            #pragma unroll
            for (int j4 = 0; j4 < CD / 4; ++j4) {        // 8x ds_load_b128, broadcast
                const v4f b4 = bp4[j4];
                #pragma unroll
                for (int c = 0; c < 4; ++c)
                    ov[j4 * 4 + c] = fmaf(av, b4[c], ov[j4 * 4 + c]);
            }
        }

        float ss = 0.0f;
        #pragma unroll
        for (int j = 0; j < CD; ++j) {
            ov[j] *= 0.25f;                              // mean over 16 * sqrt(16)
            ss = fmaf(ov[j], ov[j], ss);
        }
        #pragma unroll
        for (int off = 16; off > 0; off >>= 1)           // wave32 reduction
            ss += __shfl_xor(ss, off, 32);

        const float scale = rsqrtf(ss * (1.0f / FEAT) + 1e-12f);

        // write-once stream -> non-temporal stores
        v4f* orow = reinterpret_cast<v4f*>(out + (size_t)(t0 + tok) * FEAT + lane * CD);
        #pragma unroll
        for (int j4 = 0; j4 < CD / 4; ++j4) {
            v4f v;
            #pragma unroll
            for (int c = 0; c < 4; ++c) v[c] = ov[j4 * 4 + c] * scale;
            __builtin_nontemporal_store(v, orow + j4);
        }
    }
}

extern "C" void kernel_launch(void* const* d_in, const int* in_sizes, int n_in,
                              void* d_out, int out_size, void* d_ws, size_t ws_size,
                              hipStream_t stream) {
    const float* x = (const float*)d_in[0];   // [4, 2048, 1024] f32
    const float* w = (const float*)d_in[1];   // [16, 1024, 64]  f32
    float* out = (float*)d_out;               // [4, 2048, 1024] f32

    const int tokens = in_sizes[0] / FEAT;    // 8192
    const int grid   = tokens / TOK_TILE;     // 512

    if (ws_size >= WPREP_BYTES) {
        __bf16* whi = (__bf16*)d_ws;
        __bf16* wlo = whi + WPREP_ELEMS;
        prep_weights<<<WPREP_ELEMS / 16 / 256, 256, 0, stream>>>(w, whi, wlo);
        dynam_linear_fused<true><<<grid, BLOCK, 0, stream>>>(x, w, whi, wlo, out);
    } else {
        dynam_linear_fused<false><<<grid, BLOCK, 0, stream>>>(x, w, nullptr, nullptr, out);
    }
}
